// MoE_50663434223694
// MI455X (gfx1250) — compile-verified, matrix operand order checked by reading, not run on hardware
//
#include <hip/hip_runtime.h>
#include <hip/hip_bf16.h>

typedef __attribute__((ext_vector_type(16))) __bf16 v16bf;
typedef __attribute__((ext_vector_type(8)))  float  v8f;

#define T_TOK 8192
#define DDIM  1024
#define HDIM  2048
#define NEXP  8
#define NPAIR (T_TOK * 2)   // 16384 compact (token, expert) pairs

#define APITCH_U 20   // A-tile LDS pitch in dwords (64 rows x 16 dwords + pad)
#define BPITCH_U 28   // B-tile LDS pitch in dwords (64 n-rows x 16 kpair dwords + pad)

union FragU {
    v16bf v;
    uint4 q[2];
};

// ---------------------------------------------------------------------------
// elementwise helpers
// ---------------------------------------------------------------------------
__global__ void zero_f32_kernel(float* __restrict__ p, int n) {
    for (int i = blockIdx.x * blockDim.x + threadIdx.x; i < n; i += gridDim.x * blockDim.x)
        p[i] = 0.0f;
}

__global__ void zero_i32_kernel(int* __restrict__ p, int n) {
    int i = blockIdx.x * blockDim.x + threadIdx.x;
    if (i < n) p[i] = 0;
}

__global__ void cvt_bf16_kernel(const float* __restrict__ s, __bf16* __restrict__ d, int n) {
    for (int i = blockIdx.x * blockDim.x + threadIdx.x; i < n; i += gridDim.x * blockDim.x)
        d[i] = (__bf16)s[i];
}

// ---------------------------------------------------------------------------
// router: logits, top-2, normalized sigmoid gates, per-expert counts
// ---------------------------------------------------------------------------
__global__ void router_kernel(const float* __restrict__ x, const float* __restrict__ rw,
                              int* __restrict__ topi, float* __restrict__ topg,
                              int* __restrict__ counts) {
    int t = blockIdx.x * blockDim.x + threadIdx.x;
    if (t >= T_TOK) return;
    const float4* xr = (const float4*)(x + (size_t)t * DDIM);
    float lg[NEXP];
#pragma unroll
    for (int e = 0; e < NEXP; ++e) {
        const float4* wr = (const float4*)(rw + (size_t)e * DDIM);
        float acc = 0.0f;
        for (int d = 0; d < DDIM / 4; ++d) {
            float4 a = xr[d], b = wr[d];
            acc += a.x * b.x + a.y * b.y + a.z * b.z + a.w * b.w;
        }
        lg[e] = acc;
    }
    int i0 = 0;
#pragma unroll
    for (int e = 1; e < NEXP; ++e) if (lg[e] > lg[i0]) i0 = e;
    int i1 = (i0 == 0) ? 1 : 0;
#pragma unroll
    for (int e = 0; e < NEXP; ++e) if (e != i0 && lg[e] > lg[i1]) i1 = e;
    float g0 = __builtin_amdgcn_rcpf(1.0f + __expf(-lg[i0]));
    float g1 = __builtin_amdgcn_rcpf(1.0f + __expf(-lg[i1]));
    float inv = __builtin_amdgcn_rcpf(g0 + g1 + 1e-10f);
    topi[t * 2 + 0] = i0;  topg[t * 2 + 0] = g0 * inv;
    topi[t * 2 + 1] = i1;  topg[t * 2 + 1] = g1 * inv;
    atomicAdd(&counts[i0], 1);
    atomicAdd(&counts[i1], 1);
}

__global__ void offsets_kernel(const int* __restrict__ counts,
                               int* __restrict__ offs, int* __restrict__ cursor) {
    if (threadIdx.x == 0 && blockIdx.x == 0) {
        int run = 0;
        for (int e = 0; e < NEXP; ++e) { offs[e] = run; cursor[e] = run; run += counts[e]; }
    }
}

__global__ void scatter_kernel(const int* __restrict__ topi, const float* __restrict__ topg,
                               int* __restrict__ cursor,
                               int* __restrict__ rowmap, float* __restrict__ gatemap) {
    int t = blockIdx.x * blockDim.x + threadIdx.x;
    if (t >= T_TOK) return;
#pragma unroll
    for (int k = 0; k < 2; ++k) {
        int e = topi[t * 2 + k];
        int pos = atomicAdd(&cursor[e], 1);
        rowmap[pos]  = t;
        gatemap[pos] = topg[t * 2 + k];
    }
}

// ---------------------------------------------------------------------------
// Fragment loaders. Both A and (swizzled) B tiles are stored so that the 8
// dwords a lane needs are two contiguous 16B runs:
//   lane group kh = lane>>4:  dwords {4kh .. 4kh+3} and {8+4kh .. 11+4kh}
// A tile:  sa[m][kpair]        (natural row-major along K)
// B tile:  sb[n][kpair]        (K-pair-packed at fill time via v_perm_b32)
// ---------------------------------------------------------------------------
__device__ inline v16bf load_frag16(const unsigned* row, int kh4) {
    FragU f;
    f.q[0] = *(const uint4*)(row + kh4);
    f.q[1] = *(const uint4*)(row + 8 + kh4);
    return f.v;
}

__device__ inline float silu_mul(float g, float u) {
    return g * __builtin_amdgcn_rcpf(1.0f + __expf(-g)) * u;
}

// ---------------------------------------------------------------------------
// gate/up fused GEMM: h = silu(Xe @ Wg) * (Xe @ Wu), bf16 out
// block: 64(M rows of expert segment) x 64(N of H), 128 threads = 4 waves 2x2
// ---------------------------------------------------------------------------
__launch_bounds__(128)
__global__ void gateup_kernel(const __bf16* __restrict__ xbf,
                              const __bf16* __restrict__ wg,
                              const __bf16* __restrict__ wu,
                              const int* __restrict__ rowmap,
                              const int* __restrict__ offs,
                              const int* __restrict__ counts,
                              __bf16* __restrict__ hbuf) {
    const int e     = blockIdx.z;
    const int cnt   = counts[e];
    const int mbase = 64 * blockIdx.y;
    if (mbase >= cnt) return;
    const int row0 = offs[e] + mbase;
    const int n0   = 64 * blockIdx.x;

    __shared__ __align__(16) unsigned sx[64 * APITCH_U];
    __shared__ __align__(16) unsigned sg[64 * BPITCH_U];
    __shared__ __align__(16) unsigned su[64 * BPITCH_U];
    __shared__ int stok[64];

    const int tid  = threadIdx.x;
    const int lane = tid & 31;
    const int wave = tid >> 5;
    const int wm   = wave >> 1;   // 0..1
    const int wn   = wave & 1;    // 0..1
    const int kh4  = (lane >> 4) * 4;
    const int lmod = lane & 15;

    if (tid < 64) {
        int r = row0 + tid;
        if (r > NPAIR - 1) r = NPAIR - 1;
        stok[tid] = rowmap[r];
    }

    v8f accg[2][2], accu[2][2];
#pragma unroll
    for (int i = 0; i < 2; ++i)
#pragma unroll
        for (int j = 0; j < 2; ++j) {
            accg[i][j] = (v8f){0.f, 0.f, 0.f, 0.f, 0.f, 0.f, 0.f, 0.f};
            accu[i][j] = (v8f){0.f, 0.f, 0.f, 0.f, 0.f, 0.f, 0.f, 0.f};
        }

    const unsigned* xu  = (const unsigned*)xbf;
    const unsigned* wgu = (const unsigned*)(wg + (size_t)e * DDIM * HDIM);
    const unsigned* wuu = (const unsigned*)(wu + (size_t)e * DDIM * HDIM);

    for (int k0 = 0; k0 < DDIM; k0 += 32) {
        __syncthreads();
        // X tile: 64 rows x 16 dwords (natural layout)
#pragma unroll
        for (int i = 0; i < 8; ++i) {
            int idx = tid + i * 128;      // 0..1023
            int r = idx >> 4, u = idx & 15;
            sx[r * APITCH_U + u] = xu[(size_t)stok[r] * (DDIM / 2) + (k0 >> 1) + u];
        }
        // W tiles: read dword pairs from rows k, k+1; v_perm-pack into [n][kpair]
#pragma unroll
        for (int i = 0; i < 4; ++i) {
            int idx = tid + i * 128;      // 0..511
            int q = idx & 31;             // n-pair index (n = 2q, 2q+1)
            int p = idx >> 5;             // k-pair index 0..15
            size_t gi = (size_t)(k0 + 2 * p) * (HDIM / 2) + (n0 >> 1) + q;
            unsigned a0 = wgu[gi], a1 = wgu[gi + HDIM / 2];
            sg[(2 * q) * BPITCH_U + p]     = __builtin_amdgcn_perm(a1, a0, 0x05040100u);
            sg[(2 * q + 1) * BPITCH_U + p] = __builtin_amdgcn_perm(a1, a0, 0x07060302u);
            unsigned b0 = wuu[gi], b1 = wuu[gi + HDIM / 2];
            su[(2 * q) * BPITCH_U + p]     = __builtin_amdgcn_perm(b1, b0, 0x05040100u);
            su[(2 * q + 1) * BPITCH_U + p] = __builtin_amdgcn_perm(b1, b0, 0x07060302u);
        }
        __syncthreads();

        v16bf a[2], bg[2], bu[2];
#pragma unroll
        for (int tm = 0; tm < 2; ++tm)
            a[tm] = load_frag16(sx + (wm * 32 + tm * 16 + lmod) * APITCH_U, kh4);
#pragma unroll
        for (int tn = 0; tn < 2; ++tn) {
            bg[tn] = load_frag16(sg + (wn * 32 + tn * 16 + lmod) * BPITCH_U, kh4);
            bu[tn] = load_frag16(su + (wn * 32 + tn * 16 + lmod) * BPITCH_U, kh4);
        }
#pragma unroll
        for (int tm = 0; tm < 2; ++tm)
#pragma unroll
            for (int tn = 0; tn < 2; ++tn) {
                accg[tm][tn] = __builtin_amdgcn_wmma_f32_16x16x32_bf16(
                    false, a[tm], false, bg[tn], (short)0, accg[tm][tn], false, false);
                accu[tm][tn] = __builtin_amdgcn_wmma_f32_16x16x32_bf16(
                    false, a[tm], false, bu[tn], (short)0, accu[tm][tn], false, false);
            }
    }

    // epilogue: h = silu(g) * u  -> bf16.  Uniform fast path for full tiles.
    const int grp = lane >> 4;
    if (mbase + 64 <= cnt) {
#pragma unroll
        for (int tm = 0; tm < 2; ++tm)
#pragma unroll
            for (int tn = 0; tn < 2; ++tn) {
                int n = n0 + wn * 32 + tn * 16 + lmod;
                __bf16* dst = hbuf + (size_t)(row0 + wm * 32 + tm * 16 + 8 * grp) * HDIM + n;
#pragma unroll
                for (int r = 0; r < 8; ++r)
                    dst[(size_t)r * HDIM] = (__bf16)silu_mul(accg[tm][tn][r], accu[tm][tn][r]);
            }
    } else {
#pragma unroll
        for (int tm = 0; tm < 2; ++tm)
#pragma unroll
            for (int tn = 0; tn < 2; ++tn) {
                int n = n0 + wn * 32 + tn * 16 + lmod;
#pragma unroll
                for (int r = 0; r < 8; ++r) {
                    int m = wm * 32 + tm * 16 + r + 8 * grp;
                    if (mbase + m < cnt)
                        hbuf[(size_t)(row0 + m) * HDIM + n] =
                            (__bf16)silu_mul(accg[tm][tn][r], accu[tm][tn][r]);
                }
            }
    }
}

// ---------------------------------------------------------------------------
// down GEMM: out[token] += gate * (h @ Wd[e]),  atomic scatter-add
// ---------------------------------------------------------------------------
__launch_bounds__(128)
__global__ void down_kernel(const __bf16* __restrict__ hbuf,
                            const __bf16* __restrict__ wd,
                            const int* __restrict__ rowmap,
                            const float* __restrict__ gatemap,
                            const int* __restrict__ offs,
                            const int* __restrict__ counts,
                            float* __restrict__ out) {
    const int e     = blockIdx.z;
    const int cnt   = counts[e];
    const int mbase = 64 * blockIdx.y;
    if (mbase >= cnt) return;
    const int row0 = offs[e] + mbase;
    const int n0   = 64 * blockIdx.x;

    __shared__ __align__(16) unsigned sh[64 * APITCH_U];
    __shared__ __align__(16) unsigned sw[64 * BPITCH_U];
    __shared__ int   stok[64];
    __shared__ float sgt[64];

    const int tid  = threadIdx.x;
    const int lane = tid & 31;
    const int wave = tid >> 5;
    const int wm   = wave >> 1;
    const int wn   = wave & 1;
    const int kh4  = (lane >> 4) * 4;
    const int lmod = lane & 15;

    if (tid < 64) {
        int r = row0 + tid;
        if (r > NPAIR - 1) r = NPAIR - 1;
        stok[tid] = rowmap[r];
        sgt[tid]  = gatemap[r];
    }

    v8f acc[2][2];
#pragma unroll
    for (int i = 0; i < 2; ++i)
#pragma unroll
        for (int j = 0; j < 2; ++j)
            acc[i][j] = (v8f){0.f, 0.f, 0.f, 0.f, 0.f, 0.f, 0.f, 0.f};

    const unsigned* hu  = (const unsigned*)hbuf;
    const unsigned* wdu = (const unsigned*)(wd + (size_t)e * HDIM * DDIM);

    for (int k0 = 0; k0 < HDIM; k0 += 32) {
        __syncthreads();
#pragma unroll
        for (int i = 0; i < 8; ++i) {
            int idx = tid + i * 128;
            int r = idx >> 4, u = idx & 15;
            int src = row0 + r;
            if (src > NPAIR - 1) src = NPAIR - 1;
            sh[r * APITCH_U + u] = hu[(size_t)src * (HDIM / 2) + (k0 >> 1) + u];
        }
#pragma unroll
        for (int i = 0; i < 4; ++i) {
            int idx = tid + i * 128;      // 0..511
            int q = idx & 31;
            int p = idx >> 5;
            size_t gi = (size_t)(k0 + 2 * p) * (DDIM / 2) + (n0 >> 1) + q;
            unsigned a0 = wdu[gi], a1 = wdu[gi + DDIM / 2];
            sw[(2 * q) * BPITCH_U + p]     = __builtin_amdgcn_perm(a1, a0, 0x05040100u);
            sw[(2 * q + 1) * BPITCH_U + p] = __builtin_amdgcn_perm(a1, a0, 0x07060302u);
        }
        __syncthreads();

        v16bf a[2], b[2];
#pragma unroll
        for (int tm = 0; tm < 2; ++tm)
            a[tm] = load_frag16(sh + (wm * 32 + tm * 16 + lmod) * APITCH_U, kh4);
#pragma unroll
        for (int tn = 0; tn < 2; ++tn)
            b[tn] = load_frag16(sw + (wn * 32 + tn * 16 + lmod) * BPITCH_U, kh4);
#pragma unroll
        for (int tm = 0; tm < 2; ++tm)
#pragma unroll
            for (int tn = 0; tn < 2; ++tn)
                acc[tm][tn] = __builtin_amdgcn_wmma_f32_16x16x32_bf16(
                    false, a[tm], false, b[tn], (short)0, acc[tm][tn], false, false);
    }

    const int grp = lane >> 4;
    if (mbase + 64 <= cnt) {
#pragma unroll
        for (int tm = 0; tm < 2; ++tm)
#pragma unroll
            for (int tn = 0; tn < 2; ++tn) {
                int n = n0 + wn * 32 + tn * 16 + lmod;
#pragma unroll
                for (int r = 0; r < 8; ++r) {
                    int m = wm * 32 + tm * 16 + r + 8 * grp;
                    atomicAdd(&out[(size_t)stok[m] * DDIM + n], sgt[m] * acc[tm][tn][r]);
                }
            }
    } else {
#pragma unroll
        for (int tm = 0; tm < 2; ++tm)
#pragma unroll
            for (int tn = 0; tn < 2; ++tn) {
                int n = n0 + wn * 32 + tn * 16 + lmod;
#pragma unroll
                for (int r = 0; r < 8; ++r) {
                    int m = wm * 32 + tm * 16 + r + 8 * grp;
                    if (mbase + m < cnt)
                        atomicAdd(&out[(size_t)stok[m] * DDIM + n], sgt[m] * acc[tm][tn][r]);
                }
            }
    }
}

// ---------------------------------------------------------------------------
// launch
// ---------------------------------------------------------------------------
extern "C" void kernel_launch(void* const* d_in, const int* in_sizes, int n_in,
                              void* d_out, int out_size, void* d_ws, size_t ws_size,
                              hipStream_t stream) {
    const float* x  = (const float*)d_in[0];   // [T, D]
    const float* rw = (const float*)d_in[1];   // [E, D]
    const float* wg = (const float*)d_in[2];   // [E, D, H]
    const float* wu = (const float*)d_in[3];   // [E, D, H]
    const float* wd = (const float*)d_in[4];   // [E, H, D]
    float* out = (float*)d_out;

    // workspace partition (256B aligned slabs)
    char* ws = (char*)d_ws;
    size_t off = 0;
    auto take = [&](size_t bytes) -> void* {
        void* p = ws + off;
        off = (off + bytes + 255) & ~(size_t)255;
        return p;
    };
    __bf16* xbf   = (__bf16*)take((size_t)T_TOK * DDIM * 2);
    __bf16* wgbf  = (__bf16*)take((size_t)NEXP * DDIM * HDIM * 2);
    __bf16* wubf  = (__bf16*)take((size_t)NEXP * DDIM * HDIM * 2);
    __bf16* wdbf  = (__bf16*)take((size_t)NEXP * HDIM * DDIM * 2);
    __bf16* hbuf  = (__bf16*)take((size_t)NPAIR * HDIM * 2);
    int*    rowmap  = (int*)take(NPAIR * 4);
    float*  gatemap = (float*)take(NPAIR * 4);
    int*    topi    = (int*)take(NPAIR * 4);
    float*  topg    = (float*)take(NPAIR * 4);
    int*    counts  = (int*)take(NEXP * 4);
    int*    offs    = (int*)take(NEXP * 4);
    int*    cursor  = (int*)take(NEXP * 4);
    (void)ws_size; (void)in_sizes; (void)n_in;

    // 0) zero out + counts
    zero_f32_kernel<<<4096, 256, 0, stream>>>(out, out_size);
    zero_i32_kernel<<<1, 32, 0, stream>>>(counts, NEXP);

    // 1) fp32 -> bf16 conversions (weights become L2-resident working set)
    cvt_bf16_kernel<<<8192, 256, 0, stream>>>(x,  xbf,  T_TOK * DDIM);
    cvt_bf16_kernel<<<8192, 256, 0, stream>>>(wg, wgbf, NEXP * DDIM * HDIM);
    cvt_bf16_kernel<<<8192, 256, 0, stream>>>(wu, wubf, NEXP * DDIM * HDIM);
    cvt_bf16_kernel<<<8192, 256, 0, stream>>>(wd, wdbf, NEXP * HDIM * DDIM);

    // 2) routing + compact per-expert segments
    router_kernel<<<T_TOK / 256, 256, 0, stream>>>(x, rw, topi, topg, counts);
    offsets_kernel<<<1, 1, 0, stream>>>(counts, offs, cursor);
    scatter_kernel<<<T_TOK / 256, 256, 0, stream>>>(topi, topg, cursor, rowmap, gatemap);

    // 3) expert GEMMs (y-range covers worst-case segment length)
    dim3 gridGU(HDIM / 64, NPAIR / 64, NEXP);
    gateup_kernel<<<gridGU, 128, 0, stream>>>(xbf, wgbf, wubf, rowmap, offs, counts, hbuf);

    dim3 gridDN(DDIM / 64, NPAIR / 64, NEXP);
    down_kernel<<<gridDN, 128, 0, stream>>>(hbuf, wdbf, rowmap, gatemap, offs, counts, out);
}
